// RangeBatchNorm2d_custom_180388626598
// MI455X (gfx1250) — compile-verified, matrix operand order checked by reading, not run on hardware
//
#include <hip/hip_runtime.h>

// Problem dims (fixed by setup_inputs)
#define B_    64
#define C_    128
#define HW_   3136
#define CHW_  (C_ * HW_)          // 401408
#define K_    8                   // num_chunks
#define BPC_  (B_ / K_)           // 8 batches per chunk
#define CHUNK_ (BPC_ * HW_)       // 25088 elements per chunk
#define NTOT_  (B_ * HW_)         // 200704 elements per channel

typedef __attribute__((ext_vector_type(16))) __bf16 v16bf;
typedef __attribute__((ext_vector_type(8)))  float  v8f;
typedef __attribute__((ext_vector_type(4)))  float  f4;

// bf16 round-to-nearest-even round trip (the reference's _q)
static __device__ __forceinline__ float qbf(float x) { return (float)(__bf16)x; }

// Workspace layout (floats):
//   [0,              C_*K_)      chunk max   [c][k]
//   [C_*K_,        2*C_*K_)      chunk min   [c][k]
//   [2*C_*K_,      3*C_*K_)      chunk sum   [c][k]
//   [3*C_*K_,      3*C_*K_+C_)   avg[c]
//   [3*C_*K_+C_,   3*C_*K_+2C_)  scale[c]

// ---------------------------------------------------------------------------
// Pass 1: per-(channel, chunk) max / min / sum over 8 batches * 3136 elems.
// 1024 blocks x 256 threads (8 wave32). Pure streaming reduce -> direct
// global_load_b128 (float4); this pass also warms x into the 192MB L2.
// ---------------------------------------------------------------------------
__global__ void __launch_bounds__(256)
rbn_stats_k(const float* __restrict__ x, float* __restrict__ ws) {
    const int tid = threadIdx.x;
    const int c   = blockIdx.x >> 3;   // channel
    const int k   = blockIdx.x & 7;    // chunk
    const float* base = x + ((size_t)(k * BPC_) * C_ + c) * HW_;

    float vmax = -__builtin_inff();
    float vmin =  __builtin_inff();
    float vsum = 0.0f;

    for (int b = 0; b < BPC_; ++b) {
        const f4* p = (const f4*)(base + (size_t)b * CHW_);
        for (int i = tid; i < HW_ / 4; i += 256) {
            f4 v = p[i];
#pragma unroll
            for (int e = 0; e < 4; ++e) {
                float t = qbf(v[e]);
                vmax = fmaxf(vmax, t);
                vmin = fminf(vmin, t);
                vsum += t;
            }
        }
    }

    // wave32 butterfly reduction
#pragma unroll
    for (int off = 16; off > 0; off >>= 1) {
        vmax = fmaxf(vmax, __shfl_xor(vmax, off, 32));
        vmin = fminf(vmin, __shfl_xor(vmin, off, 32));
        vsum +=           __shfl_xor(vsum, off, 32);
    }

    __shared__ float smax[8], smin[8], ssum[8];
    const int w = tid >> 5, lane = tid & 31;
    if (lane == 0) { smax[w] = vmax; smin[w] = vmin; ssum[w] = vsum; }
    __syncthreads();

    if (tid == 0) {
        float M = smax[0], m = smin[0], s = ssum[0];
#pragma unroll
        for (int i = 1; i < 8; ++i) {
            M = fmaxf(M, smax[i]); m = fminf(m, smin[i]); s += ssum[i];
        }
        const int idx = c * K_ + k;
        ws[idx]              = M;
        ws[C_ * K_ + idx]    = m;
        ws[2 * C_ * K_ + idx] = s;
    }
}

// ---------------------------------------------------------------------------
// Pass 2: combine chunk partials -> per-channel avg & scale.
// The (128 x 8) reduce-along-chunks is done on the matrix unit:
//   D(16x16 f32) = A(16x32 bf16) x B(32x16 bf16)
// A rows   = 16 channels; A K=0..7 -> chunk maxes, K=8..15 -> chunk mins
//   (A, contiguous k order used here: lane<16 row M=lane holds K=0..15; lanes 16..31 hold K=16..31,
//    which we leave zero.)
// B selector: col 0 sums K=0..7, col 1 sums K=8..15
//   (B, same contiguous k order: lane<16 = column N=lane, 16 bf16 elems = K=0..15.)
// Chunk maxes/mins are exactly bf16-representable -> WMMA result is exact.
// Total sum (fp32, not bf16-exact) stays scalar.
// ---------------------------------------------------------------------------
__global__ void __launch_bounds__(256)
rbn_combine_k(float* __restrict__ ws) {
    const int tid  = threadIdx.x;      // 256 threads = 8 waves; wave w -> channels 16w..16w+15
    const int w    = tid >> 5;
    const int lane = tid & 31;

    const float* pmax = ws;
    const float* pmin = ws + C_ * K_;
    const float* psum = ws + 2 * C_ * K_;

    v16bf a = {};
    const int c0 = w * 16;
    if (lane < 16) {
        const int c = c0 + lane;
#pragma unroll
        for (int e = 0; e < 8; ++e) {
            a[e]     = (__bf16)pmax[c * K_ + e];
            a[8 + e] = (__bf16)pmin[c * K_ + e];
        }
    }

    v16bf b = {};
    if (lane == 0) {
#pragma unroll
        for (int e = 0; e < 8; ++e) b[e] = (__bf16)1.0f;      // col 0: sum of maxes
    }
    if (lane == 1) {
#pragma unroll
        for (int e = 0; e < 8; ++e) b[8 + e] = (__bf16)1.0f;  // col 1: sum of mins
    }

    v8f acc = {};
    acc = __builtin_amdgcn_wmma_f32_16x16x32_bf16(
        /*neg_a=*/false, a, /*neg_b=*/false, b,
        /*c_mod=*/(short)0, acc, /*reuse_a=*/false, /*reuse_b=*/false);

    // D layout: lane n (<16) VGPR r -> D[M=r][N=n]; lane 16+n VGPR r -> D[M=8+r][N=n]
    __shared__ float red[8][2][16];
    if (lane == 0)  { for (int r = 0; r < 8; ++r) red[w][0][r]     = acc[r]; }
    if (lane == 16) { for (int r = 0; r < 8; ++r) red[w][0][8 + r] = acc[r]; }
    if (lane == 1)  { for (int r = 0; r < 8; ++r) red[w][1][r]     = acc[r]; }
    if (lane == 17) { for (int r = 0; r < 8; ++r) red[w][1][8 + r] = acc[r]; }
    __syncthreads();

    if (tid < C_) {
        const int ww = tid >> 4, row = tid & 15;
        const float sum_max = qbf(red[ww][0][row]);
        const float sum_min = qbf(red[ww][1][row]);
        const float avg_max = qbf(sum_max / (float)K_);
        const float avg_min = qbf(sum_min / (float)K_);

        float tot = 0.0f;
#pragma unroll
        for (int e = 0; e < 8; ++e) tot += psum[tid * K_ + e];
        const float avg = qbf(qbf(tot) / (float)NTOT_);

        const float scale_fix = rsqrtf(2.0f * logf((float)CHUNK_));
        const float scale = qbf(1.0f / ((avg_max - avg_min) * scale_fix + 1e-5f));

        ws[3 * C_ * K_ + tid]       = avg;
        ws[3 * C_ * K_ + C_ + tid]  = scale;
    }
}

// ---------------------------------------------------------------------------
// Pass 3: elementwise normalize. x re-read hits L2 (103MB fits in 192MB);
// output stored non-temporally so the streaming writes don't evict x.
// ---------------------------------------------------------------------------
__global__ void __launch_bounds__(256)
rbn_norm_k(const float* __restrict__ x, const float* __restrict__ gamma,
           const float* __restrict__ beta, const float* __restrict__ ws,
           float* __restrict__ out) {
    const size_t i4 = (size_t)blockIdx.x * 256 + threadIdx.x;   // float4 index
    const int c = (int)((i4 / (HW_ / 4)) % C_);

    const float avg = ws[3 * C_ * K_ + c];
    const float sc  = ws[3 * C_ * K_ + C_ + c];
    const float g   = qbf(gamma[c]);
    const float bt  = beta[c];

    f4 v = ((const f4*)x)[i4];
    f4 o;
#pragma unroll
    for (int e = 0; e < 4; ++e) {
        o[e] = qbf(qbf((qbf(v[e]) - avg) * sc) * g + bt);
    }
    __builtin_nontemporal_store(o, ((f4*)out) + i4);
}

extern "C" void kernel_launch(void* const* d_in, const int* in_sizes, int n_in,
                              void* d_out, int out_size, void* d_ws, size_t ws_size,
                              hipStream_t stream) {
    const float* x     = (const float*)d_in[0];
    const float* gamma = (const float*)d_in[1];
    const float* beta  = (const float*)d_in[2];
    // d_in[3] = num_chunks (=8, fixed by setup_inputs; kernels are specialized)
    float* ws  = (float*)d_ws;
    float* out = (float*)d_out;

    rbn_stats_k<<<C_ * K_, 256, 0, stream>>>(x, ws);
    rbn_combine_k<<<1, 256, 0, stream>>>(ws);
    const int n4 = (B_ * C_ * HW_) / 4;          // 6,422,528 float4s
    rbn_norm_k<<<n4 / 256, 256, 0, stream>>>(x, gamma, beta, ws, out);
}
